// TrimodalCrossAttention_91087666414014
// MI455X (gfx1250) — matrix operand order repaired and checked by reading, not compiled, not run on hardware
//
#include <hip/hip_runtime.h>
#include <hip/hip_bf16.h>

// ---------------------------------------------------------------------------
// TrimodalCrossAttention, folded.
// seq_len==1 attention => softmax==1 => attn out = (x@Wv.T+bv)@Wo.T+bo.
// Whole net folds to: hidden = relu([g|d|c] @ A.T + beff); out = hidden@W2.T + b2
// GEMM1: [B,768]x[768,512], GEMM2: [B,512]x[512,256], both f16 WMMA, hidden in LDS.
// Weight k-blocks streamed with global_load_async_to_lds_b128 (ASYNCcnt,
// double-buffered) so fetch of block k+1 overlaps WMMA of block k.
// ---------------------------------------------------------------------------

typedef __attribute__((ext_vector_type(16))) _Float16 v16h;
typedef __attribute__((ext_vector_type(8)))  _Float16 v8h;
typedef __attribute__((ext_vector_type(4)))  _Float16 v4h;
typedef __attribute__((ext_vector_type(8)))  float    v8f;

#define E 256
#define BROWS 131072
#define MTILE 64
#define XS_STRIDE 776   // 768 + 8 halves pad (16B aligned rows, bank-conflict free)
#define HS_STRIDE 520   // 512 + 8
#define BS_STRIDE 264   // 256 + 8
#define BSBUF (32 * BS_STRIDE)   // one B buffer (halves)

// ---------------- precompute kernels (tiny) ----------------

// M[i] = Wo_i @ Wv_i   (6 x 256 x 256)
__global__ void k_fold_M(const float* __restrict__ Wo, const float* __restrict__ Wv,
                         float* __restrict__ M) {
  int idx = blockIdx.x * 256 + threadIdx.x;     // 6*256*256
  int i = idx >> 16; int nk = idx & 0xFFFF; int n = nk >> 8; int k = nk & 255;
  const float* wo = Wo + (size_t)i * 65536 + (size_t)n * 256;
  const float* wv = Wv + (size_t)i * 65536;
  float s = 0.f;
  for (int t = 0; t < 256; ++t) s += wo[t] * wv[t * 256 + k];
  M[idx] = s;
}

// cvec[i][n] = Wo_i[n,:].bv_i + bo_i[n]
__global__ void k_fold_c(const float* __restrict__ Wo, const float* __restrict__ bv,
                         const float* __restrict__ bo, float* __restrict__ cvec) {
  int idx = blockIdx.x * 256 + threadIdx.x;     // 6*256
  int i = idx >> 8; int n = idx & 255;
  const float* wo = Wo + (size_t)i * 65536 + (size_t)n * 256;
  const float* b  = bv + (size_t)i * 256;
  float s = bo[i * 256 + n];
  for (int t = 0; t < 256; ++t) s += wo[t] * b[t];
  cvec[idx] = s;
}

// AT[k][n] (f16, K-major [768,512]) = folded first-layer weight, pre-transposed
// for direct WMMA B-fragment loads.
__global__ void k_build_AT(const float* __restrict__ W1, const float* __restrict__ M,
                           _Float16* __restrict__ AT) {
  int idx = blockIdx.x * 256 + threadIdx.x;     // 768*512
  int k = idx >> 9; int n = idx & 511;
  int b = k >> 8; int kk = k & 255;
  const float* w1 = W1 + (size_t)n * 768;
  float s = w1[k];
  int mi0, off0, mi1, off1;
  if (b == 0)      { mi0 = 2; off0 = 256; mi1 = 4; off1 = 512; }  // g col
  else if (b == 1) { mi0 = 0; off0 = 0;   mi1 = 5; off1 = 512; }  // d col
  else             { mi0 = 1; off0 = 0;   mi1 = 3; off1 = 256; }  // c col
  const float* m0 = M + (size_t)mi0 * 65536;
  const float* m1 = M + (size_t)mi1 * 65536;
  for (int t = 0; t < 256; ++t)
    s += w1[off0 + t] * m0[t * 256 + kk] + w1[off1 + t] * m1[t * 256 + kk];
  AT[(size_t)k * 512 + n] = (_Float16)s;
}

// beff[n] = b1[n] + W1g[n,:]. (c0+c1) + W1d[n,:]. (c2+c3) + W1c[n,:]. (c4+c5)
__global__ void k_build_beff(const float* __restrict__ W1, const float* __restrict__ b1,
                             const float* __restrict__ cvec, float* __restrict__ beff) {
  int n = blockIdx.x * 256 + threadIdx.x;       // 512
  const float* w1 = W1 + (size_t)n * 768;
  float s = b1[n];
  for (int t = 0; t < 256; ++t) {
    s += w1[t]       * (cvec[0 * 256 + t] + cvec[1 * 256 + t]);
    s += w1[256 + t] * (cvec[2 * 256 + t] + cvec[3 * 256 + t]);
    s += w1[512 + t] * (cvec[4 * 256 + t] + cvec[5 * 256 + t]);
  }
  beff[n] = s;
}

// W2T[k][n] = W2[n][k]  (f16, [512,256] K-major)
__global__ void k_build_W2T(const float* __restrict__ W2, _Float16* __restrict__ W2T) {
  int idx = blockIdx.x * 256 + threadIdx.x;     // 512*256
  int k = idx >> 8; int n = idx & 255;
  W2T[idx] = (_Float16)W2[(size_t)n * 512 + k];
}

// ---------------- fused main kernel ----------------

static __device__ __forceinline__ v16h cat8(v8h lo, v8h hi) {
  return __builtin_shufflevector(lo, hi, 0, 1, 2, 3, 4, 5, 6, 7,
                                 8, 9, 10, 11, 12, 13, 14, 15);
}

// Async cache->LDS copy of 16 bytes per lane (GLOBAL_LOAD_ASYNC_TO_LDS_B128,
// GV mode: 64-bit VGPR address, saddr=off). Tracked by ASYNCcnt.
// Low 32 bits of a generic LDS pointer are the LDS byte offset (aperture rule).
static __device__ __forceinline__ void async_copy16(_Float16* dst,
                                                    const _Float16* src) {
  unsigned      lds = (unsigned)(uintptr_t)dst;
  unsigned long long ga = (unsigned long long)(uintptr_t)src;
  asm volatile("global_load_async_to_lds_b128 %0, %1, off"
               :: "v"(lds), "v"(ga) : "memory");
}

static __device__ __forceinline__ void wait_async0() {
  asm volatile("s_wait_asynccnt 0x0" ::: "memory");
}

__global__ __launch_bounds__(256) void k_fused(
    const float* __restrict__ G, const float* __restrict__ D,
    const float* __restrict__ C,
    const _Float16* __restrict__ AT, const float* __restrict__ beff,
    const _Float16* __restrict__ W2T, const float* __restrict__ b2,
    float* __restrict__ out) {
  extern __shared__ _Float16 smem[];
  _Float16* Xs = smem;                         // [64][XS_STRIDE]
  _Float16* Hs = smem + MTILE * XS_STRIDE;     // [64][HS_STRIDE]
  _Float16* Bs = Hs + MTILE * HS_STRIDE;       // 2 x [32][BS_STRIDE]

  const int t    = threadIdx.x;
  const int wg   = blockIdx.x;
  const int lane = t & 31;
  const int wave = t >> 5;
  const int r    = wave & 3;               // row-tile within M-tile
  const int ng   = wave >> 2;              // N-group (0/1)
  const int mrow = lane & 15;
  const int hi8  = (lane >> 4) << 3;       // C/D: lanes 16-31 hold M+8
  const int koff = (lane < 16) ? 0 : 8;    // A frag: lanes 16-31 hold K+8 block

  // per-thread staging coordinates (32x256 B block, 64B per thread)
  const int skk = t >> 3;
  const int snl = (t & 7) * 32;

  // ---- stage X tile [64 x 768] as f16 into LDS ----
  {
    const int row0 = wg * MTILE;
    for (int i = 0; i < 48; ++i) {
      int c = t + i * 256;                 // 0..12287
      int row = c / 192;
      int k4  = (c % 192) * 4;
      const float* src;
      if (k4 < 256)      src = G + (size_t)(row0 + row) * E + k4;
      else if (k4 < 512) src = D + (size_t)(row0 + row) * E + (k4 - 256);
      else               src = C + (size_t)(row0 + row) * E + (k4 - 512);
      float4 f = *(const float4*)src;
      v4h h;
      h[0] = (_Float16)f.x; h[1] = (_Float16)f.y;
      h[2] = (_Float16)f.z; h[3] = (_Float16)f.w;
      *(v4h*)(&Xs[row * XS_STRIDE + k4]) = h;
    }
  }
  __syncthreads();

  // ---- GEMM1: hidden[64,512] = relu(X @ A.T + beff), kept in LDS ----
  for (int nc = 0; nc < 2; ++nc) {
    v8f acc[8];
#pragma unroll
    for (int i = 0; i < 8; ++i) { v8f z = {0, 0, 0, 0, 0, 0, 0, 0}; acc[i] = z; }

    const int nglob = (snl < 128) ? (nc * 128 + snl)
                                  : (256 + nc * 128 + (snl - 128));
    // prologue: async-stage k-block 0 into buffer 0
    {
      const _Float16* src = AT + (size_t)skk * 512 + nglob;
      _Float16* dst = &Bs[skk * BS_STRIDE + snl];
#pragma unroll
      for (int q = 0; q < 4; ++q) async_copy16(dst + q * 8, src + q * 8);
    }
    wait_async0();
    __syncthreads();

    for (int ks = 0; ks < 24; ++ks) {
      const int buf = ks & 1;
      // async-stage next k-block into the other buffer (overlaps WMMA below)
      if (ks + 1 < 24) {
        const _Float16* src = AT + (size_t)((ks + 1) * 32 + skk) * 512 + nglob;
        _Float16* dst = &Bs[(buf ^ 1) * BSBUF + skk * BS_STRIDE + snl];
#pragma unroll
        for (int q = 0; q < 4; ++q) async_copy16(dst + q * 8, src + q * 8);
      }

      const int kb = ks * 32;
      const _Float16* ap = &Xs[(16 * r + mrow) * XS_STRIDE + kb + koff];
      v16h a = cat8(*(const v8h*)ap, *(const v8h*)(ap + 16));

#pragma unroll
      for (int nt = 0; nt < 8; ++nt) {
        const _Float16* bp = &Bs[buf * BSBUF + koff * BS_STRIDE + ng * 128 + nt * 16 + mrow];
        v16h b;
#pragma unroll
        for (int s = 0; s < 8; ++s) { b[s] = bp[s * BS_STRIDE]; b[s + 8] = bp[(s + 16) * BS_STRIDE]; }
        acc[nt] = __builtin_amdgcn_wmma_f32_16x16x32_f16(
            false, a, false, b, (short)0, acc[nt], false, false);
      }
      wait_async0();        // own async fills of buf^1 have landed
      __syncthreads();      // everyone done reading buf / filling buf^1
    }

    // epilogue: bias + relu -> f16 hidden in LDS
#pragma unroll
    for (int nt = 0; nt < 8; ++nt) {
      int nh = ng * 256 + nc * 128 + nt * 16 + mrow;
      float bias = beff[nh];
#pragma unroll
      for (int v = 0; v < 8; ++v) {
        float x = acc[nt][v] + bias;
        x = x > 0.f ? x : 0.f;
        Hs[(16 * r + v + hi8) * HS_STRIDE + nh] = (_Float16)x;
      }
    }
    __syncthreads();
  }

  // ---- GEMM2: out[64,256] = hidden @ W2.T + b2 ----
  v8f acc2[8];
#pragma unroll
  for (int i = 0; i < 8; ++i) { v8f z = {0, 0, 0, 0, 0, 0, 0, 0}; acc2[i] = z; }

  // prologue: stage k-block 0 into buffer 0
  {
    const _Float16* src = W2T + (size_t)skk * 256 + snl;
    _Float16* dst = &Bs[skk * BS_STRIDE + snl];
#pragma unroll
    for (int q = 0; q < 4; ++q) async_copy16(dst + q * 8, src + q * 8);
  }
  wait_async0();
  __syncthreads();

  for (int ks = 0; ks < 16; ++ks) {
    const int buf = ks & 1;
    if (ks + 1 < 16) {
      const _Float16* src = W2T + (size_t)((ks + 1) * 32 + skk) * 256 + snl;
      _Float16* dst = &Bs[(buf ^ 1) * BSBUF + skk * BS_STRIDE + snl];
#pragma unroll
      for (int q = 0; q < 4; ++q) async_copy16(dst + q * 8, src + q * 8);
    }

    const int kb = ks * 32;
    const _Float16* ap = &Hs[(16 * r + mrow) * HS_STRIDE + kb + koff];
    v16h a = cat8(*(const v8h*)ap, *(const v8h*)(ap + 16));

#pragma unroll
    for (int nt = 0; nt < 8; ++nt) {
      const _Float16* bp = &Bs[buf * BSBUF + koff * BS_STRIDE + ng * 128 + nt * 16 + mrow];
      v16h b;
#pragma unroll
      for (int s = 0; s < 8; ++s) { b[s] = bp[s * BS_STRIDE]; b[s + 8] = bp[(s + 16) * BS_STRIDE]; }
      acc2[nt] = __builtin_amdgcn_wmma_f32_16x16x32_f16(
          false, a, false, b, (short)0, acc2[nt], false, false);
    }
    wait_async0();
    __syncthreads();
  }

  // epilogue -> f32 output
#pragma unroll
  for (int nt = 0; nt < 8; ++nt) {
    int n = ng * 128 + nt * 16 + mrow;
    float bias = b2[n];
#pragma unroll
    for (int v = 0; v < 8; ++v) {
      int mg = wg * MTILE + 16 * r + v + hi8;
      out[(size_t)mg * E + n] = acc2[nt][v] + bias;
    }
  }
}

// ---------------- host launcher ----------------

extern "C" void kernel_launch(void* const* d_in, const int* in_sizes, int n_in,
                              void* d_out, int out_size, void* d_ws, size_t ws_size,
                              hipStream_t stream) {
  (void)in_sizes; (void)n_in; (void)out_size; (void)ws_size;
  const float* G  = (const float*)d_in[0];
  const float* Dd = (const float*)d_in[1];
  const float* Cc = (const float*)d_in[2];
  // d_in[3],[4] (Wq,Wk) and d_in[6],[7] (bq,bk) are mathematically dead:
  // softmax over a single key is identically 1.
  const float* Wv = (const float*)d_in[5];
  const float* bv = (const float*)d_in[8];
  const float* Wo = (const float*)d_in[9];
  const float* bo = (const float*)d_in[10];
  const float* W1 = (const float*)d_in[11];
  const float* b1 = (const float*)d_in[12];
  const float* W2 = (const float*)d_in[13];
  const float* b2 = (const float*)d_in[14];
  float* out = (float*)d_out;

  char* ws = (char*)d_ws;
  float*    Mfold = (float*)ws;     ws += (size_t)6 * 256 * 256 * 4;
  float*    cvec  = (float*)ws;     ws += (size_t)6 * 256 * 4;
  float*    beff  = (float*)ws;     ws += (size_t)512 * 4;
  _Float16* AT    = (_Float16*)ws;  ws += (size_t)768 * 512 * 2;
  _Float16* W2T   = (_Float16*)ws;  ws += (size_t)512 * 256 * 2;

  k_fold_M    <<<1536, 256, 0, stream>>>(Wo, Wv, Mfold);
  k_fold_c    <<<6,    256, 0, stream>>>(Wo, bv, bo, cvec);
  k_build_AT  <<<1536, 256, 0, stream>>>(W1, Mfold, AT);
  k_build_beff<<<2,    256, 0, stream>>>(W1, b1, cvec, beff);
  k_build_W2T <<<512,  256, 0, stream>>>(W2, W2T);

  size_t lds = (size_t)(MTILE * XS_STRIDE + MTILE * HS_STRIDE + 2 * BSBUF) *
               sizeof(_Float16);  // ~195 KB, fits 320 KB WGP LDS
  k_fused<<<BROWS / MTILE, 256, lds, stream>>>(G, Dd, Cc, AT, beff, W2T, b2, out);
}
